// NTXentMergedMultiGPU_44607530336758
// MI455X (gfx1250) — matrix operand order repaired and checked by reading, not compiled, not run on hardware
//
#include <hip/hip_runtime.h>
#include <math.h>

// ---- problem constants (fixed by the reference) ----
constexpr int N     = 8192;   // 2*total_batch
constexpr int D     = 256;    // embedding dim
constexpr int BHALF = 4096;   // total_batch
// loss uses exp(sim / 0.5) = exp(2*sim)

constexpr int ROWS_WG    = 32;            // rows per workgroup (2 tiles of 16)
constexpr int COLS_CHUNK = 64;            // columns per chunk (4 tiles of 16)
constexpr int NCHUNK     = N / COLS_CHUNK;
constexpr int A_ELEMS    = ROWS_WG * D;   // 8192 bf16 per component
constexpr int B_ELEMS    = COLS_CHUNK * D;// 16384 bf16 per component

typedef float  v8f  __attribute__((ext_vector_type(8)));
typedef __bf16 v8bf __attribute__((ext_vector_type(8)));
typedef __bf16 v16bf __attribute__((ext_vector_type(16)));
typedef int    v4i  __attribute__((vector_size(16)));
typedef __attribute__((address_space(1))) v4i* gv4i_p;
typedef __attribute__((address_space(3))) v4i* lv4i_p;

#if __has_builtin(__builtin_amdgcn_global_load_async_to_lds_b128)
#define ASYNC_STAGE 1
#else
#define ASYNC_STAGE 0
#endif

// Copy 16 bytes (8 bf16) global -> LDS.
__device__ __forceinline__ void stage16(const __bf16* g, __bf16* l) {
#if ASYNC_STAGE
    __builtin_amdgcn_global_load_async_to_lds_b128(
        (gv4i_p)(g), (lv4i_p)(l), 0, 0);
#else
    *(float4*)l = *(const float4*)g;
#endif
}

__device__ __forceinline__ void wait_staged_le(int n) {
#if ASYNC_STAGE
 #if __has_builtin(__builtin_amdgcn_s_wait_asynccnt)
    if (n == 0) __builtin_amdgcn_s_wait_asynccnt(0);
    else        __builtin_amdgcn_s_wait_asynccnt(16);
 #else
    if (n == 0) asm volatile("s_wait_asynccnt 0x0" ::: "memory");
    else        asm volatile("s_wait_asynccnt 0x10" ::: "memory");
 #endif
#endif
}

// ---------------------------------------------------------------------------
// Kernel 1: L2-normalize each row, split into bf16 hi + lo (x ~= hi + lo)
// ---------------------------------------------------------------------------
__global__ void ntxent_normalize_split(const float* __restrict__ x,
                                       __bf16* __restrict__ zhi,
                                       __bf16* __restrict__ zlo) {
    const int row = blockIdx.x;
    const int t   = threadIdx.x;           // 256 threads == D
    const float v = x[row * D + t];
    float s = v * v;
    #pragma unroll
    for (int m = 1; m < 32; m <<= 1) s += __shfl_xor(s, m, 32);
    __shared__ float wsum[8];
    __shared__ float inv_norm;
    const int wid = t >> 5, lane = t & 31;
    if (lane == 0) wsum[wid] = s;
    __syncthreads();
    if (t == 0) {
        float tot = 0.f;
        #pragma unroll
        for (int i = 0; i < 8; ++i) tot += wsum[i];
        inv_norm = 1.0f / fmaxf(sqrtf(tot), 1e-12f);
    }
    __syncthreads();
    const float zn = v * inv_norm;
    const __bf16 h = (__bf16)zn;
    zhi[row * D + t] = h;
    zlo[row * D + t] = (__bf16)(zn - (float)h);
}

// ---------------------------------------------------------------------------
// Kernel 2: sim = z z^T via bf16x3 split on v_wmma_f32_16x16x32_bf16 with 3
//           independent accumulator chains, fused exp-row-sum + positive
//           extraction. WG = 32 rows x all 8192 cols, B chunks double-
//           buffered with async global->LDS staging. LDS = exactly 160 KB.
// ---------------------------------------------------------------------------
__global__ void ntxent_sim_rowsum(const __bf16* __restrict__ zhi,
                                  const __bf16* __restrict__ zlo,
                                  float* __restrict__ denom,
                                  float* __restrict__ pos) {
    extern __shared__ __bf16 smem[];
    __bf16* Ahi  = smem;                         // 32x256 (16 KB)
    __bf16* Alo  = Ahi + A_ELEMS;                // 32x256 (16 KB)
    __bf16* Bbuf = Alo + A_ELEMS;                // 2 x (hi+lo) 64x256 (128 KB)
    float*  rowSums = (float*)Bbuf;              // aliased: used after loop only

    const int t    = threadIdx.x;                // 256 threads = 8 waves
    const int lane = t & 31;
    const int w    = t >> 5;
    const int tr   = w >> 2;                     // row tile    0..1
    const int tc   = w & 3;                      // column tile 0..3
    const int r0   = blockIdx.x * ROWS_WG;

    // ---- stage A tile (hi & lo) + B chunk 0 ----
    #pragma unroll
    for (int i = 0; i < A_ELEMS / 8 / 256; ++i) {   // 4 per thread per comp
        const int e = 8 * (t + 256 * i);
        stage16(zhi + (size_t)r0 * D + e, Ahi + e);
        stage16(zlo + (size_t)r0 * D + e, Alo + e);
    }
    #pragma unroll
    for (int i = 0; i < B_ELEMS / 8 / 256; ++i) {   // 8 per thread per comp
        const int e = 8 * (t + 256 * i);
        stage16(zhi + e, Bbuf + e);                 // chunk 0 hi
        stage16(zlo + e, Bbuf + B_ELEMS + e);       // chunk 0 lo
    }

    // Fragment addressing (ISA 7.12.2, 16-bit, wave32, h = lane>>4):
    //   A lane (m=lane&15): K = {kb+8h .. +7} ++ {kb+16+8h .. +7}
    //   B lane (n=lane&15): K = {kb+8h .. +7} ++ {kb+16+8h .. +7}
    const int mrow = lane & 15;
    const int h    = lane >> 4;
    const int offA1 = 8 * h, offA2 = 16 + 8 * h;
    const int offB1 = 8 * h, offB2 = 16 + 8 * h;
    const __bf16* aHiRow = Ahi + (16 * tr + mrow) * D;
    const __bf16* aLoRow = aHiRow + A_ELEMS;
    const int rowTileBase = r0 + 16 * tr;

    // running exp-sums per C-tile row (valid on lanes 0 and 16 after reduce)
    float regSum[8] = {0.f, 0.f, 0.f, 0.f, 0.f, 0.f, 0.f, 0.f};

    for (int c = 0; c < NCHUNK; ++c) {
        // issue async staging for chunk c+1 into the other buffer
        if (c + 1 < NCHUNK) {
            __bf16* Bn = Bbuf + ((c + 1) & 1) * 2 * B_ELEMS;
            const size_t g0 = (size_t)(c + 1) * COLS_CHUNK * D;
            #pragma unroll
            for (int i = 0; i < B_ELEMS / 8 / 256; ++i) {
                const int e = 8 * (t + 256 * i);
                stage16(zhi + g0 + e, Bn + e);
                stage16(zlo + g0 + e, Bn + B_ELEMS + e);
            }
            wait_staged_le(16);   // chunk c complete; c+1 (16 ops) in flight
        } else {
            wait_staged_le(0);
        }
        __syncthreads();

        const __bf16* Bc = Bbuf + (c & 1) * 2 * B_ELEMS;
        const __bf16* bHiRow = Bc + (16 * tc + mrow) * D;
        const __bf16* bLoRow = bHiRow + B_ELEMS;

        // 3 independent WMMA chains (hi*hi, hi*lo, lo*hi) -> no RAW stalls
        v8f acc0 = {0.f, 0.f, 0.f, 0.f, 0.f, 0.f, 0.f, 0.f};
        v8f acc1 = acc0, acc2 = acc0;
        #pragma unroll
        for (int kb = 0; kb < D; kb += 32) {
            v8bf a0 = *(const v8bf*)(aHiRow + kb + offA1);
            v8bf a1 = *(const v8bf*)(aHiRow + kb + offA2);
            v8bf l0 = *(const v8bf*)(aLoRow + kb + offA1);
            v8bf l1 = *(const v8bf*)(aLoRow + kb + offA2);
            v8bf b0 = *(const v8bf*)(bHiRow + kb + offB1);
            v8bf b1 = *(const v8bf*)(bHiRow + kb + offB2);
            v8bf m0 = *(const v8bf*)(bLoRow + kb + offB1);
            v8bf m1 = *(const v8bf*)(bLoRow + kb + offB2);
            v16bf ahi = __builtin_shufflevector(a0, a1, 0,1,2,3,4,5,6,7,8,9,10,11,12,13,14,15);
            v16bf alo = __builtin_shufflevector(l0, l1, 0,1,2,3,4,5,6,7,8,9,10,11,12,13,14,15);
            v16bf bhi = __builtin_shufflevector(b0, b1, 0,1,2,3,4,5,6,7,8,9,10,11,12,13,14,15);
            v16bf blo = __builtin_shufflevector(m0, m1, 0,1,2,3,4,5,6,7,8,9,10,11,12,13,14,15);
            // x*y ~= hi*hi + hi*lo + lo*hi  (lo*lo ~ 2^-18, dropped)
            acc0 = __builtin_amdgcn_wmma_f32_16x16x32_bf16(false, ahi, false, bhi, (short)0, acc0, false, false);
            acc1 = __builtin_amdgcn_wmma_f32_16x16x32_bf16(false, ahi, false, blo, (short)0, acc1, false, false);
            acc2 = __builtin_amdgcn_wmma_f32_16x16x32_bf16(false, alo, false, bhi, (short)0, acc2, false, false);
        }

        // Epilogue. C layout: VGPR v -> row M = v + 8*h, col = lane&15.
        const int colTileBase = c * COLS_CHUNK + 16 * tc;
        const bool diagTile = (rowTileBase == colTileBase);
        const bool posTile  = ((rowTileBase ^ BHALF) == colTileBase);
        #pragma unroll
        for (int v = 0; v < 8; ++v) {
            const float s = acc0[v] + acc1[v] + acc2[v];
            float e = expf(2.0f * s);
            if (diagTile | posTile) {
                const int i = rowTileBase + v + 8 * h;
                const int j = colTileBase + (lane & 15);
                if (posTile && ((i ^ j) == BHALF)) pos[i] = s; // unique writer
                if (diagTile && (i == j)) e = 0.0f;            // mask diagonal
            }
            #pragma unroll
            for (int m = 1; m < 16; m <<= 1) e += __shfl_xor(e, m, 32);
            regSum[v] += e;   // group-sum present in every lane of the half
        }
        __syncthreads();   // all reads of this buffer done before reuse
    }

    // Cross-wave (tc) combination via LDS aliased onto the dead B buffer.
    if ((lane & 15) == 0) {
        #pragma unroll
        for (int v = 0; v < 8; ++v)
            rowSums[tc * ROWS_WG + 16 * tr + v + 8 * h] = regSum[v];
    }
    __syncthreads();
    if (t < ROWS_WG) {
        denom[r0 + t] = rowSums[t] + rowSums[ROWS_WG + t] +
                        rowSums[2 * ROWS_WG + t] + rowSums[3 * ROWS_WG + t];
    }
}

// ---------------------------------------------------------------------------
// Kernel 3: loss = sum_i [ log(denom_i) - 2*pos_i ] / BHALF
// ---------------------------------------------------------------------------
__global__ void ntxent_finalize(const float* __restrict__ denom,
                                const float* __restrict__ pos,
                                float* __restrict__ out) {
    const int t = threadIdx.x;
    float partial = 0.0f;
    for (int i = t; i < N; i += 256)
        partial += logf(denom[i]) - 2.0f * pos[i];
    #pragma unroll
    for (int m = 1; m < 32; m <<= 1) partial += __shfl_xor(partial, m, 32);
    __shared__ float wsum[8];
    const int wid = t >> 5, lane = t & 31;
    if (lane == 0) wsum[wid] = partial;
    __syncthreads();
    if (t == 0) {
        float tot = 0.f;
        #pragma unroll
        for (int i = 0; i < 8; ++i) tot += wsum[i];
        out[0] = tot * (1.0f / (float)BHALF);
    }
}

// ---------------------------------------------------------------------------
extern "C" void kernel_launch(void* const* d_in, const int* in_sizes, int n_in,
                              void* d_out, int out_size, void* d_ws, size_t ws_size,
                              hipStream_t stream) {
    const float* emb = (const float*)d_in[0];
    __bf16* zhi  = (__bf16*)d_ws;                    // N*D bf16 (4 MB)
    __bf16* zlo  = zhi + (size_t)N * D;              // N*D bf16 (4 MB)
    float*  denom = (float*)(zlo + (size_t)N * D);   // N floats
    float*  pos   = denom + N;                       // N floats
    float*  out   = (float*)d_out;

    ntxent_normalize_split<<<N, 256, 0, stream>>>(emb, zhi, zlo);

    const size_t smem = (size_t)(2 * A_ELEMS + 4 * B_ELEMS) * sizeof(__bf16); // 160 KB
    ntxent_sim_rowsum<<<N / ROWS_WG, 256, smem, stream>>>(zhi, zlo, denom, pos);

    ntxent_finalize<<<1, 256, 0, stream>>>(denom, pos, out);
}
